// SocialLSTM_24524263260712
// MI455X (gfx1250) — matrix operand order repaired and checked by reading, not compiled, not run on hardware
//
#include <hip/hip_runtime.h>

#define H 64
#define NGATES 256      // 4*H
#define GXC 4
#define GYC 4
#define NCELLS 16
#define M_TILE 16

typedef __attribute__((ext_vector_type(16))) _Float16 v16h;
typedef __attribute__((ext_vector_type(8)))  _Float16 v8h;
typedef __attribute__((ext_vector_type(8)))  float    v8f;

__device__ __forceinline__ float sig_(float x) { return 1.0f / (1.0f + __expf(-x)); }

// ---------------- one-shot init: zero states, f16 weights, fused bias ----------------
__global__ void sl_init(const float* __restrict__ W_hh,
                        const float* __restrict__ b_ih,
                        const float* __restrict__ b_hh,
                        _Float16* __restrict__ h_f16,
                        float* __restrict__ c_state,
                        _Float16* __restrict__ whh_f16,
                        float* __restrict__ bias,
                        float* __restrict__ tstate,
                        int n_agents)
{
    int idx = blockIdx.x * blockDim.x + threadIdx.x;
    int total = n_agents * H;
    if (idx < total) { h_f16[idx] = (_Float16)0.0f; c_state[idx] = 0.0f; }
    if (idx < NGATES * H) whh_f16[idx] = (_Float16)W_hh[idx];
    if (idx < NGATES)     bias[idx] = b_ih[idx] + b_hh[idx];
    if (idx < 2 * H)      tstate[idx] = 0.0f;
}

__global__ void sl_zero_grid(float* __restrict__ grid)
{
    int idx = blockIdx.x * blockDim.x + threadIdx.x;
    if (idx < NCELLS * H) grid[idx] = 0.0f;
}

// ---------------- per-step: N-agent LSTM (WMMA GEMM) + social-grid pooling ----------------
__global__ __launch_bounds__(256) void sl_others_step(
    const float* __restrict__ traj_target,   // (T,1,2)
    const float* __restrict__ traj_others,   // (T,N,2)
    const float* __restrict__ W_ih,          // (256,2)
    const float* __restrict__ bias,          // (256) = b_ih+b_hh
    const _Float16* __restrict__ whh_f16,    // (256,64) row-major [gate][k] == B [N][K]
    _Float16* __restrict__ h_f16,            // (N,64) state
    float* __restrict__ c_state,             // (N,64) state
    float* __restrict__ grid_out,            // (16,64)
    int t, int n_agents)
{
    __shared__ float lds_gates[M_TILE][NGATES];
    __shared__ float lds_grid[NCELLS * H];

    const int tid  = threadIdx.x;
    const int lane = tid & 31;
    const int wave = tid >> 5;
    const int agent0 = blockIdx.x * M_TILE;

    for (int p = tid; p < NCELLS * H; p += 256) lds_grid[p] = 0.0f;

    // ---- A operand: h tile (16 x 64) f16, contiguous K order per lane half (same as B) ----
    // lane (m = lane&15, hi = lane>>4): v16h elems 0..15 = K hi*16..+15 (contiguous, matches B)
    const int mrow = lane & 15;
    const int hi   = lane >> 4;
    const _Float16* arow = h_f16 + (size_t)(agent0 + mrow) * H;

    v16h a0, a1;
    {
        v8h lo0 = *(const v8h*)(arow + hi * 16);
        v8h hi0 = *(const v8h*)(arow + 8 + hi * 16);
        v8h lo1 = *(const v8h*)(arow + 32 + hi * 16);
        v8h hi1 = *(const v8h*)(arow + 40 + hi * 16);
#pragma unroll
        for (int q = 0; q < 8; ++q) {
            a0[q] = lo0[q]; a0[8 + q] = hi0[q];
            a1[q] = lo1[q]; a1[8 + q] = hi1[q];
        }
    }

    // ---- 2 column-tiles per wave, K=64 -> 2 chained WMMAs each ----
#pragma unroll
    for (int s = 0; s < 2; ++s) {
        const int ntile = wave * 2 + s;
        const int ncol  = ntile * 16 + (lane & 15);
        // B layout: lane = column, 16 contiguous K halves; K-base = (lane>>4)*16 (+32 for 2nd)
        const _Float16* brow = whh_f16 + (size_t)ncol * H + (lane >> 4) * 16;
        v16h b0 = *(const v16h*)(brow);
        v16h b1 = *(const v16h*)(brow + 32);

        v8f c = {};
        c = __builtin_amdgcn_wmma_f32_16x16x32_f16(false, a0, false, b0, (short)0, c, false, false);
        c = __builtin_amdgcn_wmma_f32_16x16x32_f16(false, a1, false, b1, (short)0, c, false, false);

        // C/D layout: elem r -> (M = (lane>>4)*8 + r, N = lane&15)
#pragma unroll
        for (int r = 0; r < 8; ++r)
            lds_gates[(lane >> 4) * 8 + r][ntile * 16 + (lane & 15)] = c[r];
    }
    __syncthreads();

    // ---- elementwise LSTM update + social-grid binning ----
    const float tpx = traj_target[t * 2 + 0];
    const float tpy = traj_target[t * 2 + 1];
    const float half_nb = 2.0f;     // NB/2
    const float inv_cw  = 1.0f;     // GX/NB
    const float inv_ch  = 1.0f;     // GY/NB

#pragma unroll
    for (int p = tid; p < M_TILE * H; p += 256) {
        const int a = p >> 6;
        const int j = p & 63;
        const int agent = agent0 + a;
        const size_t xoff = ((size_t)t * n_agents + agent) * 2;
        const float x0 = traj_others[xoff + 0];
        const float x1 = traj_others[xoff + 1];

        const int n_i = 0 * H + j, n_f = 1 * H + j, n_g = 2 * H + j, n_o = 3 * H + j;
        float gi = lds_gates[a][n_i] + W_ih[n_i * 2] * x0 + W_ih[n_i * 2 + 1] * x1 + bias[n_i];
        float gf = lds_gates[a][n_f] + W_ih[n_f * 2] * x0 + W_ih[n_f * 2 + 1] * x1 + bias[n_f];
        float gg = lds_gates[a][n_g] + W_ih[n_g * 2] * x0 + W_ih[n_g * 2 + 1] * x1 + bias[n_g];
        float go = lds_gates[a][n_o] + W_ih[n_o * 2] * x0 + W_ih[n_o * 2 + 1] * x1 + bias[n_o];

        const size_t soff = (size_t)agent * H + j;
        float c2 = sig_(gf) * c_state[soff] + sig_(gi) * tanhf(gg);
        float h2 = sig_(go) * tanhf(c2);
        c_state[soff] = c2;
        h_f16[soff] = (_Float16)h2;

        const float rx = x0 - tpx, ry = x1 - tpy;
        const bool inside = (fabsf(rx) <= half_nb) && (fabsf(ry) <= half_nb);
        const int cx = (int)(rx * inv_cw) + (GXC / 2);   // trunc toward zero, matches .long()
        const int cy = (int)(ry * inv_ch) + (GYC / 2);
        if (inside && cx >= 0 && cx < GXC && cy >= 0 && cy < GYC)
            atomicAdd(&lds_grid[(cy * GXC + cx) * H + j], h2);
    }
    __syncthreads();

    for (int p = tid; p < NCELLS * H; p += 256) {
        float v = lds_grid[p];
        if (v != 0.0f) atomicAdd(&grid_out[p], v);
    }
}

// ---------------- per-step: target LSTM + social MLP + output (1 block) ----------------
__global__ __launch_bounds__(256) void sl_target_step(
    const float* __restrict__ traj_target,
    const float* __restrict__ W_ih, const float* __restrict__ W_hh,
    const float* __restrict__ bias,
    const float* __restrict__ W1, const float* __restrict__ b1,
    const float* __restrict__ W2, const float* __restrict__ b2,
    const float* __restrict__ Wout, const float* __restrict__ bout,
    const float* __restrict__ grid,     // (16*64)
    float* __restrict__ tstate,         // h_t(64) | c_t(64)
    float* __restrict__ out, int t)
{
    __shared__ float g_lds[NGATES];
    __shared__ float hnew[H];
    __shared__ float r1[H];
    __shared__ float sc[H];
    const int tid = threadIdx.x;
    const float x0 = traj_target[t * 2 + 0];
    const float x1 = traj_target[t * 2 + 1];

    if (tid < NGATES) {
        float acc = bias[tid] + W_ih[tid * 2] * x0 + W_ih[tid * 2 + 1] * x1;
        const float* wrow = W_hh + (size_t)tid * H;
#pragma unroll 8
        for (int k = 0; k < H; ++k) acc += wrow[k] * tstate[k];
        g_lds[tid] = acc;
    }
    __syncthreads();
    if (tid < H) {
        float gi = g_lds[tid], gf = g_lds[H + tid], gg = g_lds[2 * H + tid], go = g_lds[3 * H + tid];
        float c2 = sig_(gf) * tstate[H + tid] + sig_(gi) * tanhf(gg);
        float h2 = sig_(go) * tanhf(c2);
        tstate[tid] = h2; tstate[H + tid] = c2; hnew[tid] = h2;
    }
    __syncthreads();
    if (tid < H) {
        float acc = b1[tid];
        const float* wrow = W1 + (size_t)tid * (NCELLS * H);
#pragma unroll 8
        for (int k = 0; k < NCELLS * H; ++k) acc += wrow[k] * grid[k];
        r1[tid] = fmaxf(acc, 0.0f);
    }
    __syncthreads();
    if (tid < H) {
        float acc = b2[tid];
        const float* wrow = W2 + (size_t)tid * H;
#pragma unroll 8
        for (int k = 0; k < H; ++k) acc += wrow[k] * r1[k];
        sc[tid] = acc;
    }
    __syncthreads();
    if (tid < 2) {
        float acc = bout[tid];
        const float* wrow = Wout + (size_t)tid * H;
#pragma unroll 8
        for (int k = 0; k < H; ++k) acc += wrow[k] * (hnew[k] + sc[k]);
        out[t * 2 + tid] = acc;
    }
}

extern "C" void kernel_launch(void* const* d_in, const int* in_sizes, int n_in,
                              void* d_out, int out_size, void* d_ws, size_t ws_size,
                              hipStream_t stream)
{
    const float* traj_target = (const float*)d_in[0];
    const float* traj_others = (const float*)d_in[1];
    const float* W_ih  = (const float*)d_in[2];
    const float* W_hh  = (const float*)d_in[3];
    const float* b_ih  = (const float*)d_in[4];
    const float* b_hh  = (const float*)d_in[5];
    const float* W1    = (const float*)d_in[6];
    const float* b1    = (const float*)d_in[7];
    const float* W2    = (const float*)d_in[8];
    const float* b2    = (const float*)d_in[9];
    const float* Wout  = (const float*)d_in[10];
    const float* bout  = (const float*)d_in[11];
    float* out = (float*)d_out;

    const int T = in_sizes[0] / 2;                 // 14
    const int n_agents = in_sizes[1] / (T * 2);    // 65536

    // workspace carve-out (256B aligned)
    char* ws = (char*)d_ws;
    size_t off = 0;
    auto take = [&](size_t bytes) { void* p = ws + off; off = (off + bytes + 255) & ~(size_t)255; return p; };
    _Float16* h_f16   = (_Float16*)take((size_t)n_agents * H * sizeof(_Float16));
    float*    c_state = (float*)   take((size_t)n_agents * H * sizeof(float));
    _Float16* whh_f16 = (_Float16*)take((size_t)NGATES * H * sizeof(_Float16));
    float*    bias    = (float*)   take(NGATES * sizeof(float));
    float*    grid    = (float*)   take(NCELLS * H * sizeof(float));
    float*    tstate  = (float*)   take(2 * H * sizeof(float));
    (void)ws_size;

    const int total = n_agents * H;
    sl_init<<<(total + 255) / 256, 256, 0, stream>>>(W_hh, b_ih, b_hh, h_f16, c_state,
                                                     whh_f16, bias, tstate, n_agents);
    for (int t = 0; t < T; ++t) {
        sl_zero_grid<<<(NCELLS * H + 255) / 256, 256, 0, stream>>>(grid);
        sl_others_step<<<n_agents / M_TILE, 256, 0, stream>>>(
            traj_target, traj_others, W_ih, bias, whh_f16, h_f16, c_state, grid, t, n_agents);
        sl_target_step<<<1, 256, 0, stream>>>(
            traj_target, W_ih, W_hh, bias, W1, b1, W2, b2, Wout, bout, grid, tstate, out, t);
    }
}
